// Spidercnn_cls_feature_73461120631035
// MI455X (gfx1250) — matrix operand order repaired and checked by reading, not compiled, not run on hardware
//
#include <hip/hip_runtime.h>
#include <hip/hip_bf16.h>

typedef __attribute__((ext_vector_type(16))) _Float16 v16h;
typedef __attribute__((ext_vector_type(8)))  _Float16 v8h;
typedef __attribute__((ext_vector_type(8)))  float    v8f;
typedef __attribute__((ext_vector_type(4)))  unsigned int u32x4;
typedef __attribute__((ext_vector_type(8)))  int i32x8;
typedef __attribute__((ext_vector_type(4)))  int i32x4;

#define NPT    2048
#define BATCH  4
#define KNN    20
#define NTERMS 20
#define CATCH  480

// ---------------------------------------------------------------------------
// Tensor Data Mover: 2-D tile (tile0 x tile1 elements, row stride stride0)
// from global -> LDS, packed contiguously.  6-arg builtin (clang-23 form);
// groups 2/3 zero (<=2D tile).  data_size code: 2 = 4-byte elements.
// ---------------------------------------------------------------------------
__device__ static inline void tdm_load_2d(unsigned lds_off, const void* gptr,
                                          unsigned dim0, unsigned dim1,
                                          unsigned tile0, unsigned tile1,
                                          unsigned stride0)
{
    unsigned long long ga = (unsigned long long)(uintptr_t)gptr;
    u32x4 g0;
    g0[0] = 1u;                                          // count=1, user mode
    g0[1] = lds_off;                                     // LDS byte address
    g0[2] = (unsigned)(ga & 0xFFFFFFFFu);                // global_addr[31:0]
    g0[3] = (unsigned)((ga >> 32) & 0x01FFFFFFu) | 0x80000000u;  // [56:32] | type=2
    i32x8 g1;
    g1[0] = (int)(2u << 16);                             // data_size=4B; mask/flags=0
    g1[1] = (int)((dim0 & 0xFFFFu) << 16);               // tensor_dim0[15:0]
    g1[2] = (int)((dim0 >> 16) | ((dim1 & 0xFFFFu) << 16));  // dim0 hi | dim1 lo
    g1[3] = (int)((dim1 >> 16) | (tile0 << 16));         // dim1 hi | tile_dim0
    g1[4] = (int)(tile1 & 0xFFFFu);                      // tile_dim1 (tile_dim2=0)
    g1[5] = (int)stride0;                                // tensor_dim0_stride lo32
    g1[6] = 0;                                           // stride0 hi | stride1 lo
    g1[7] = 0;
    i32x4 z4 = {0, 0, 0, 0};
    i32x8 z8 = {0, 0, 0, 0, 0, 0, 0, 0};
    __builtin_amdgcn_tensor_load_to_lds(g0, g1, z4, z4, z8, 0);
}

// ---------------------------------------------------------------------------
// KNN: one block per (b, i). d2 row in LDS, 20 argmin sweeps (low-index ties
// first, matching jax.lax.top_k ordering on -d2).
// ---------------------------------------------------------------------------
__global__ void knn_kernel(const float* __restrict__ pc, int* __restrict__ idx)
{
    __shared__ float d2[NPT];
    __shared__ float rv[256];
    __shared__ int   ri[256];
    const int b = blockIdx.x / NPT;
    const int i = blockIdx.x % NPT;
    const int tid = threadIdx.x;
    const float xi = pc[((size_t)b*NPT + i)*6 + 0];
    const float yi = pc[((size_t)b*NPT + i)*6 + 1];
    const float zi = pc[((size_t)b*NPT + i)*6 + 2];
    const float sqi = xi*xi + yi*yi + zi*zi;
    for (int j = tid; j < NPT; j += 256) {
        float xj = pc[((size_t)b*NPT + j)*6 + 0];
        float yj = pc[((size_t)b*NPT + j)*6 + 1];
        float zj = pc[((size_t)b*NPT + j)*6 + 2];
        float sqj = xj*xj + yj*yj + zj*zj;
        d2[j] = sqi + sqj - 2.0f*(xi*xj + yi*yj + zi*zj);   // same formula as reference
    }
    __syncthreads();
    for (int s = 0; s < KNN; ++s) {
        float best = 3.4e38f; int bi = NPT;
        for (int j = tid; j < NPT; j += 256) {
            float v = d2[j];
            if (v < best) { best = v; bi = j; }   // j increasing -> earliest tie kept
        }
        rv[tid] = best; ri[tid] = bi;
        __syncthreads();
        for (int st = 128; st > 0; st >>= 1) {
            if (tid < st) {
                float ov = rv[tid+st]; int oi = ri[tid+st];
                if (ov < rv[tid] || (ov == rv[tid] && oi < ri[tid])) { rv[tid] = ov; ri[tid] = oi; }
            }
            __syncthreads();
        }
        if (tid == 0) { idx[((size_t)b*NPT + i)*KNN + s] = ri[0]; d2[ri[0]] = 3.4e38f; }
        __syncthreads();
    }
}

// ---------------------------------------------------------------------------
// pc (B,N,6) -> feat0 (B,6,N) as f16
// ---------------------------------------------------------------------------
__global__ void feat0_kernel(const float* __restrict__ pc, _Float16* __restrict__ feat0)
{
    int t = blockIdx.x*blockDim.x + threadIdx.x;
    const int total = BATCH*6*NPT;
    if (t >= total) return;
    int b = t / (6*NPT);
    int c = (t / NPT) % 6;
    int n = t % NPT;
    feat0[t] = (_Float16)pc[((size_t)b*NPT + n)*6 + c];
}

// ---------------------------------------------------------------------------
// Taylor terms + per-layer taylor = w1*terms + b1.  tay layout [l][b][t][n][k].
// ---------------------------------------------------------------------------
__global__ void taylor_kernel(const float* __restrict__ pc, const int* __restrict__ idx,
                              const float* __restrict__ w1_0, const float* __restrict__ b1_0,
                              const float* __restrict__ w1_1, const float* __restrict__ b1_1,
                              const float* __restrict__ w1_2, const float* __restrict__ b1_2,
                              const float* __restrict__ w1_3, const float* __restrict__ b1_3,
                              float* __restrict__ tay)
{
    int t = blockIdx.x*blockDim.x + threadIdx.x;
    const int total = BATCH*NPT*KNN;
    if (t >= total) return;
    int b = t / (NPT*KNN);
    int n = (t / KNN) % NPT;
    int j = idx[t];
    float xi = pc[((size_t)b*NPT + n)*6 + 0];
    float yi = pc[((size_t)b*NPT + n)*6 + 1];
    float zi = pc[((size_t)b*NPT + n)*6 + 2];
    float X = pc[((size_t)b*NPT + j)*6 + 0] - xi;
    float Y = pc[((size_t)b*NPT + j)*6 + 1] - yi;
    float Z = pc[((size_t)b*NPT + j)*6 + 2] - zi;
    float tm[NTERMS];
    tm[0]=1.0f; tm[1]=X; tm[2]=Y; tm[3]=Z; tm[4]=X*Y; tm[5]=X*Z; tm[6]=Y*Z;
    tm[7]=X*X; tm[8]=Y*Y; tm[9]=Z*Z; tm[10]=X*X*Y; tm[11]=X*X*Z; tm[12]=X*Y*Y;
    tm[13]=Y*Y*Z; tm[14]=X*Z*Z; tm[15]=Y*Z*Z; tm[16]=X*Y*Z;
    tm[17]=X*X*X; tm[18]=Y*Y*Y; tm[19]=Z*Z*Z;
    const float* w1s[4] = {w1_0, w1_1, w1_2, w1_3};
    const float* b1s[4] = {b1_0, b1_1, b1_2, b1_3};
    int n_k = t % (NPT*KNN);   // n*KNN + k
    #pragma unroll
    for (int l = 0; l < 4; ++l) {
        #pragma unroll
        for (int tt = 0; tt < 3; ++tt) {
            float s = b1s[l][tt];
            #pragma unroll
            for (int u = 0; u < NTERMS; ++u) s += tm[u]*w1s[l][tt*NTERMS + u];
            tay[(((size_t)l*BATCH + b)*3 + tt)*((size_t)NPT*KNN) + n_k] = s;
        }
    }
}

// ---------------------------------------------------------------------------
// w2 (Cout, CK) f32 -> f16, rows padded to CK32 (multiple of 32) with zeros.
// ---------------------------------------------------------------------------
__global__ void convw_kernel(const float* __restrict__ w2, _Float16* __restrict__ Wh,
                             int Cout, int CK, int CK32)
{
    int i = blockIdx.x*blockDim.x + threadIdx.x;
    if (i >= Cout*CK32) return;
    int o = i / CK32, q = i % CK32;
    Wh[i] = (q < CK) ? (_Float16)w2[(size_t)o*CK + q] : (_Float16)0.0f;
}

// ---------------------------------------------------------------------------
// Decode LUT: q -> (c<<7)|(t<<5)|k for q<CK, -1 in the zero pad.
// ---------------------------------------------------------------------------
__global__ void lut_kernel(int* __restrict__ lut, int CK, int CK32)
{
    int q = blockIdx.x*blockDim.x + threadIdx.x;
    if (q >= CK32) return;
    int r = -1;
    if (q < CK) {
        int k  = q % KNN;
        int cc = q / KNN;
        int tt = cc % 3;
        int c  = cc / 3;
        r = (c << 7) | (tt << 5) | k;
    }
    lut[q] = r;
}

// ---------------------------------------------------------------------------
// Fused SpiderConv GEMM: Out(Cout,16 pts) = W(Cout,CK) x Ahat(CK,16 pts).
// One workgroup per 16-point tile; one wave per 16-output-channel M-tile.
// idx/taylor/lut tiles DMA'd to LDS by the Tensor Data Mover; Ahat chunk
// double-buffered so the gather/builder for chunk i+1 overlaps the WMMA on
// chunk i.  Branch-free builder for all full chunks (HASPAD only layer 0).
// ---------------------------------------------------------------------------
template<int WAVES, int CK32, bool HASPAD>
__global__ void __launch_bounds__(WAVES*32)
spider_conv_kernel(const _Float16* __restrict__ featIn,   // (B, Cin, N) f16
                   const _Float16* __restrict__ Wh,       // (Cout, CK32) f16
                   const float*    __restrict__ tay,      // (B, 3, N, K) this layer
                   const int*      __restrict__ idx,      // (B, N, K)
                   const int*      __restrict__ lut,      // (CK32) decode table
                   const float*    __restrict__ b2,       // (Cout)
                   _Float16*       __restrict__ featOut,  // (B, Cout, N) f16
                   float*          __restrict__ catOut,   // (B, 480, N) f32
                   int Cin, int Cout, int chOff)
{
    constexpr int THREADS = WAVES*32;
    constexpr int BSTRIDE = 40;   // 80B row stride: 16 cols -> 16 distinct banks, 16B aligned
    __shared__ int   idxLds[16*KNN];
    __shared__ float tayLds[3*16*KNN];
    __shared__ int   lutLds[CK32];
    __shared__ alignas(64) _Float16 Bc[2][16*BSTRIDE];

    const int tile = blockIdx.x;
    const int b    = tile / (NPT/16);
    const int n0   = (tile % (NPT/16)) * 16;
    const int tid  = threadIdx.x;
    const int lane = tid & 31;
    const int m0   = (tid >> 5) * 16;            // this wave's M-tile

    const int*      idxB  = idx + ((size_t)b*NPT + n0)*KNN;
    const float*    tayB  = tay + ((size_t)b*3*NPT + n0)*(size_t)KNN;
    const _Float16* featB = featIn + (size_t)b*Cin*NPT;

    // ---- TDM: DMA tile-invariant idx / taylor / lut into LDS (wave 0 only) ----
    if (tid < 32) {
        // idx tile: 320 contiguous ints
        tdm_load_2d((unsigned)(uintptr_t)(void*)idxLds, idxB,
                    16*KNN, 1, 16*KNN, 1, 16*KNN);
        // taylor tile: 3 rows of 320 contiguous floats, row stride NPT*KNN
        tdm_load_2d((unsigned)(uintptr_t)(void*)tayLds, tayB,
                    16*KNN, 3, 16*KNN, 3, NPT*KNN);
        // decode LUT: CK32 contiguous ints
        tdm_load_2d((unsigned)(uintptr_t)(void*)lutLds, lut,
                    CK32, 1, CK32, 1, CK32);
        __builtin_amdgcn_s_wait_tensorcnt(0);
    }
    __syncthreads();

    // ---- Ahat chunk builders: 512 f16 values, statically unrolled ----
    auto buildFull = [&](int q0, int buf) {   // every q < CK: no branch
        #pragma unroll
        for (int it = 0; it < 512/THREADS; ++it) {
            int v   = tid + it*THREADS;
            int col = v >> 5;
            int kk  = v & 31;
            int lv  = lutLds[q0 + kk];
            int k  = lv & 31;
            int tt = (lv >> 5) & 3;
            int c  = lv >> 7;
            int j  = idxLds[col*KNN + k];
            float ty = tayLds[(tt*16 + col)*KNN + k];
            float f  = (float)featB[(size_t)c*NPT + j];   // the only global gather
            Bc[buf][col*BSTRIDE + kk] = (_Float16)(f * ty);
        }
    };
    auto buildTail = [&](int q0, int buf) {   // padded chunk (layer 0 only)
        #pragma unroll
        for (int it = 0; it < 512/THREADS; ++it) {
            int v   = tid + it*THREADS;
            int col = v >> 5;
            int kk  = v & 31;
            int lv  = lutLds[q0 + kk];
            float val = 0.0f;
            if (lv >= 0) {
                int k  = lv & 31;
                int tt = (lv >> 5) & 3;
                int c  = lv >> 7;
                int j  = idxLds[col*KNN + k];
                val = (float)featB[(size_t)c*NPT + j] * tayLds[(tt*16 + col)*KNN + k];
            }
            Bc[buf][col*BSTRIDE + kk] = (_Float16)val;
        }
    };

    buildFull(0, 0);
    __syncthreads();

    const int mlane = m0 + (lane & 15);
    const int offA  = (lane < 16) ? 0 : 8;    // A 16x32 f16: K runs {0-7,16-23}/{8-15,24-31}
    const int colr  = lane & 15;              // B 32x16 f16: col=lane, K {0-7,16-23}/{8-15,24-31}
    const int offB  = (lane < 16) ? 0 : 8;

    v8f acc = {};
    int buf = 0;
    #pragma unroll 1
    for (int q0 = 0; q0 < CK32; q0 += 32) {
        // A fragment: two 16B runs of this wave's W rows (L2-resident)
        const _Float16* wp = Wh + (size_t)mlane*CK32 + q0 + offA;
        __builtin_prefetch(wp + 32, 0, 1);
        v8h wlo = *(const v8h*)(wp);
        v8h whi = *(const v8h*)(wp + 16);
        v16h afrag;
        #pragma unroll
        for (int i2 = 0; i2 < 8; ++i2) { afrag[i2] = wlo[i2]; afrag[8+i2] = whi[i2]; }
        // B fragment from current LDS buffer
        const v8h* bp = (const v8h*)&Bc[buf][colr*BSTRIDE + offB];
        v8h blo = bp[0];
        v8h bhi = bp[2];
        v16h bfrag;
        #pragma unroll
        for (int i2 = 0; i2 < 8; ++i2) { bfrag[i2] = blo[i2]; bfrag[8+i2] = bhi[i2]; }
        // overlap: build next chunk into the other buffer
        const int qn = q0 + 32;
        if (qn < CK32) {
            if (HASPAD && qn == CK32 - 32) buildTail(qn, buf ^ 1);
            else                           buildFull(qn, buf ^ 1);
        }
        acc = __builtin_amdgcn_wmma_f32_16x16x32_f16(
            false, afrag, false, bfrag, (short)0, acc, false, false);
        __syncthreads();
        buf ^= 1;
    }

    // ---- epilogue: bias + relu; D layout: VGPR r -> M=r (lanes 0-15) / r+8 ----
    const int mbase = m0 + ((lane < 16) ? 0 : 8);
    const int n     = n0 + colr;
    #pragma unroll
    for (int r = 0; r < 8; ++r) {
        int m = mbase + r;
        float v = acc[r] + b2[m];
        v = v > 0.0f ? v : 0.0f;
        featOut[((size_t)b*Cout + m)*NPT + n] = (_Float16)v;
        catOut[((size_t)b*CATCH + chOff + m)*NPT + n] = v;
    }
}

// ---------------------------------------------------------------------------
// top-2 over N per (b, channel) row of cat.
// ---------------------------------------------------------------------------
__global__ void top2_kernel(const float* __restrict__ cat, float* __restrict__ out)
{
    __shared__ float s1[256], s2[256];
    const int row = blockIdx.x;                 // b*480 + ch
    const float* p = cat + (size_t)row*NPT;
    const int tid = threadIdx.x;
    float m1 = -3.4e38f, m2 = -3.4e38f;
    for (int j = tid; j < NPT; j += 256) {
        float v = p[j];
        if (v > m1) { m2 = m1; m1 = v; } else if (v > m2) m2 = v;
    }
    s1[tid] = m1; s2[tid] = m2;
    __syncthreads();
    for (int st = 128; st > 0; st >>= 1) {
        if (tid < st) {
            float a1 = s1[tid], a2 = s2[tid], c1 = s1[tid+st], c2 = s2[tid+st];
            float n1 = a1 > c1 ? a1 : c1;
            float lo = a1 > c1 ? c1 : a1;
            float h2 = a2 > c2 ? a2 : c2;
            s1[tid] = n1;
            s2[tid] = lo > h2 ? lo : h2;
        }
        __syncthreads();
    }
    if (tid == 0) {
        int b = row / CATCH, ch = row % CATCH;
        out[(size_t)b*(CATCH*2) + ch*2 + 0] = s1[0];
        out[(size_t)b*(CATCH*2) + ch*2 + 1] = s2[0];
    }
}

// ---------------------------------------------------------------------------
extern "C" void kernel_launch(void* const* d_in, const int* in_sizes, int n_in,
                              void* d_out, int out_size, void* d_ws, size_t ws_size,
                              hipStream_t stream)
{
    (void)in_sizes; (void)n_in; (void)out_size; (void)ws_size;
    const float* pc = (const float*)d_in[0];
    const float* w1[4] = {(const float*)d_in[1], (const float*)d_in[5],
                          (const float*)d_in[9], (const float*)d_in[13]};
    const float* b1[4] = {(const float*)d_in[2], (const float*)d_in[6],
                          (const float*)d_in[10], (const float*)d_in[14]};
    const float* w2[4] = {(const float*)d_in[3], (const float*)d_in[7],
                          (const float*)d_in[11], (const float*)d_in[15]};
    const float* b2[4] = {(const float*)d_in[4], (const float*)d_in[8],
                          (const float*)d_in[12], (const float*)d_in[16]};
    float* out = (float*)d_out;

    const int Cin[4]  = {6, 32, 64, 128};
    const int Cout[4] = {32, 64, 128, 256};
    int CK[4], CK32[4];
    for (int l = 0; l < 4; ++l) { CK[l] = Cin[l]*3*KNN; CK32[l] = (CK[l] + 31) & ~31; }

    char* ws = (char*)d_ws;
    size_t off = 0;
    auto alloc = [&](size_t bytes) -> char* {
        char* p = ws + off;
        off += (bytes + 255) & ~(size_t)255;
        return p;
    };
    int*      idx   = (int*)     alloc((size_t)BATCH*NPT*KNN*4);
    float*    tay   = (float*)   alloc((size_t)4*BATCH*3*NPT*KNN*4);
    _Float16* featA = (_Float16*)alloc((size_t)BATCH*256*NPT*2);
    _Float16* featB = (_Float16*)alloc((size_t)BATCH*256*NPT*2);
    float*    cat   = (float*)   alloc((size_t)BATCH*CATCH*NPT*4);
    _Float16* Wh[4];
    int*      lut[4];
    for (int l = 0; l < 4; ++l) Wh[l]  = (_Float16*)alloc((size_t)Cout[l]*CK32[l]*2);
    for (int l = 0; l < 4; ++l) lut[l] = (int*)alloc((size_t)CK32[l]*4);

    feat0_kernel<<<(BATCH*6*NPT + 255)/256, 256, 0, stream>>>(pc, featA);
    knn_kernel<<<BATCH*NPT, 256, 0, stream>>>(pc, idx);
    taylor_kernel<<<(BATCH*NPT*KNN + 255)/256, 256, 0, stream>>>(pc, idx,
        w1[0], b1[0], w1[1], b1[1], w1[2], b1[2], w1[3], b1[3], tay);
    for (int l = 0; l < 4; ++l) {
        int tot = Cout[l]*CK32[l];
        convw_kernel<<<(tot + 255)/256, 256, 0, stream>>>(w2[l], Wh[l], Cout[l], CK[l], CK32[l]);
        lut_kernel<<<(CK32[l] + 255)/256, 256, 0, stream>>>(lut[l], CK[l], CK32[l]);
    }
    const int chOff[4] = {0, 32, 96, 224};
    const size_t tayStride = (size_t)BATCH*3*NPT*KNN;
    _Float16* fin = featA; _Float16* fout = featB;
    const int grid = BATCH*(NPT/16);
    for (int l = 0; l < 4; ++l) {
        const float* tl = tay + (size_t)l*tayStride;
        switch (l) {
        case 0: spider_conv_kernel<2, 384, true><<<grid, 2*32, 0, stream>>>(
                    fin, Wh[l], tl, idx, lut[l], b2[l], fout, cat, Cin[l], Cout[l], chOff[l]); break;
        case 1: spider_conv_kernel<4, 1920, false><<<grid, 4*32, 0, stream>>>(
                    fin, Wh[l], tl, idx, lut[l], b2[l], fout, cat, Cin[l], Cout[l], chOff[l]); break;
        case 2: spider_conv_kernel<8, 3840, false><<<grid, 8*32, 0, stream>>>(
                    fin, Wh[l], tl, idx, lut[l], b2[l], fout, cat, Cin[l], Cout[l], chOff[l]); break;
        default: spider_conv_kernel<16, 7680, false><<<grid, 16*32, 0, stream>>>(
                    fin, Wh[l], tl, idx, lut[l], b2[l], fout, cat, Cin[l], Cout[l], chOff[l]); break;
        }
        _Float16* tmp = fin; fin = fout; fout = tmp;
    }
    top2_kernel<<<BATCH*CATCH, 256, 0, stream>>>(cat, out);
}
